// BiGRU_76338748719479
// MI455X (gfx1250) — compile-verified, matrix operand order checked by reading, not run on hardware
//
#include <hip/hip_runtime.h>
#include <hip/hip_bf16.h>
#include <math.h>

// Problem constants
#define V_  50000
#define E_  256
#define H_  512
#define B_  128
#define T_  512
#define G3  1536          // 3*H
#define BT_ (B_*T_)
#define HP  520           // padded bf16 LDS row stride for h (breaks bank aliasing)

typedef __attribute__((ext_vector_type(16))) __bf16 v16bf;
typedef __attribute__((ext_vector_type(8)))  float  v8f;
typedef __attribute__((ext_vector_type(4)))  unsigned int u32x4;
typedef __attribute__((ext_vector_type(4)))  float  f32x4;

union Frag {            // one lane's 16 bf16 (32B) WMMA A/B fragment
    v16bf  v;
    __bf16 h[16];
    u32x4  u[2];
};
union AccF {            // one lane's 8 f32 C/D fragment
    v8f   v;
    float f[8];
};

__device__ __forceinline__ float fsigmoid(float x) {
    // 1/(1+e^-x) via v_exp_f32 + v_rcp_f32
    return __builtin_amdgcn_rcpf(1.f + __builtin_amdgcn_exp2f(-1.4426950408889634f * x));
}
__device__ __forceinline__ float ftanh(float x) {
    // tanh(x) = 2*sigmoid(2x) - 1
    return 2.f * __builtin_amdgcn_rcpf(1.f + __builtin_amdgcn_exp2f(-2.8853900817779268f * x)) - 1.f;
}

// ---------------------------------------------------------------------------
// Pack a row-major f32 weight W[Gdim][K] (gemm B-matrix is W^T: (k,n)=W[n][k])
// into per-lane WMMA bf16 B-fragments. Fragment index = nt*(K/32)+kc; lane l
// holds column n = nt*16+(l&15); lanes<16 get K {kc*32+0..7, +16..23},
// lanes>=16 get K {kc*32+8..15, +24..31}  (CDNA5 16-bit operand layout).
// Fragment is 32 lanes * 32B = 1KB; lane's 32B slice at frag*1024 + lane*32.
// ---------------------------------------------------------------------------
__global__ void pack_w_bf16(const float* __restrict__ W, __bf16* __restrict__ P,
                            int Gdim, int K) {
    int tid  = blockIdx.x * blockDim.x + threadIdx.x;
    int KC   = K >> 5;
    int nTil = (Gdim >> 4) * KC;
    int tile = tid >> 5;
    int lane = tid & 31;
    if (tile >= nTil) return;
    int nt = tile / KC, kc = tile % KC;
    int n     = nt * 16 + (lane & 15);
    int kbase = kc * 32 + ((lane < 16) ? 0 : 8);
    const float* wr = W + (size_t)n * K + kbase;
    Frag f;
#pragma unroll
    for (int i = 0; i < 8; i++) f.h[i]     = (__bf16)wr[i];
#pragma unroll
    for (int i = 0; i < 8; i++) f.h[8 + i] = (__bf16)wr[16 + i];
    u32x4* dst = (u32x4*)(P + ((size_t)tile * 32 + lane) * 16);
    dst[0] = f.u[0];
    dst[1] = f.u[1];
}

// ---------------------------------------------------------------------------
// Fused embedding-gather + input projection:
//   xp[dir][bt][g] = sum_e emb[inputs[bt]][e] * W_ih[dir][g][e] + b_ih[dir][g]
// One block = one 16-row M-tile of BT; 8 waves. Waves 0-3 -> forward dir,
// waves 4-7 -> backward dir, 24 N-tiles each, processed 4 at a time: four
// independent accumulator chains share the register-resident A-fragments,
// and all B-fragment loads use one base pointer + constant offsets so the
// scheduler can clause/prefetch them arbitrarily far ahead.
// ---------------------------------------------------------------------------
__global__ __launch_bounds__(256)
void xproj_kernel(const int*   __restrict__ inputs,
                  const float* __restrict__ emb,
                  const __bf16* __restrict__ PWf,
                  const __bf16* __restrict__ PWb,
                  const float* __restrict__ bihf,
                  const float* __restrict__ bihb,
                  __bf16* __restrict__ xpF,
                  __bf16* __restrict__ xpB) {
    int wave = threadIdx.x >> 5;
    int lane = threadIdx.x & 31;
    int m0   = blockIdx.x * 16;
    int row  = m0 + (lane & 15);
    int tok  = inputs[row];
    const float* xr = emb + (size_t)tok * E_;
    int koff = (lane < 16) ? 0 : 8;

    Frag a[8];                               // full K=256 of A in registers
#pragma unroll
    for (int kc = 0; kc < 8; kc++) {
        const float* p = xr + kc * 32 + koff;
#pragma unroll
        for (int i = 0; i < 8; i++) a[kc].h[i]     = (__bf16)p[i];
#pragma unroll
        for (int i = 0; i < 8; i++) a[kc].h[8 + i] = (__bf16)p[16 + i];
    }

    int dir = (wave >= 4);
    const __bf16* PW   = dir ? PWb  : PWf;
    const float*  bias = dir ? bihb : bihf;
    __bf16*       xp   = dir ? xpB  : xpF;
    int base  = (wave & 3) * 24;             // 24 local N-tiles per wave
    int rbase = m0 + ((lane < 16) ? 0 : 8);  // D: VGPR r -> M=r / M=8+r

    // wave's weight base: fragment (tile base+jt, kc) at elem (jt*8+kc)*512
    const __bf16* wb0 = PW + (size_t)base * 8 * 512 + lane * 16;

#pragma unroll 1
    for (int it = 0; it < 6; it++) {
        const __bf16* wb = wb0 + (size_t)it * 4 * 8 * 512;
        AccF acc[4];
#pragma unroll
        for (int j = 0; j < 4; j++)
#pragma unroll
            for (int i = 0; i < 8; i++) acc[j].f[i] = 0.f;

#pragma unroll
        for (int kc = 0; kc < 8; kc++) {
#pragma unroll
            for (int j = 0; j < 4; j++) {
                Frag b;
                const u32x4* bp = (const u32x4*)(wb + (size_t)(j * 8 + kc) * 512);
                b.u[0] = bp[0];
                b.u[1] = bp[1];
                acc[j].v = __builtin_amdgcn_wmma_f32_16x16x32_bf16(
                    false, a[kc].v, false, b.v, (short)0, acc[j].v, false, false);
            }
        }
#pragma unroll
        for (int j = 0; j < 4; j++) {
            int   ntl = base + it * 4 + j;
            int   col = ntl * 16 + (lane & 15);
            float bv  = bias[col];
#pragma unroll
            for (int r = 0; r < 8; r++)
                __builtin_nontemporal_store((__bf16)(acc[j].f[r] + bv),
                                            xp + (size_t)(rbase + r) * G3 + col);
        }
    }
}

// ---------------------------------------------------------------------------
// Persistent recurrent scan. Grid = 16 WGs (2 dirs x 8 batch slices of 16
// rows), 512 threads = 16 waves. Hidden state lives in LDS for all T steps:
// f32 copy (update math) + padded bf16 copy (WMMA A operand). Per step each
// wave runs ONE K-sweep with SIX independent accumulator chains (its 6
// N-tiles of hg = h @ W_hh^T): each LDS A-fragment feeds 6 WMMAs, and all 12
// B-fragment loads per kc come from one base pointer + constant offsets, so
// the W_hh L2 stream (the roofline limiter) is fully pipelined. Gate
// pre-activations staged in LDS; combine phase vectorized (wave=row,
// lane=16-col chunk), xp read non-temporally (streamed once, 400MB).
// ---------------------------------------------------------------------------
__global__ __launch_bounds__(512, 1)
void gru_scan_kernel(const __bf16* __restrict__ PWhhF,
                     const __bf16* __restrict__ PWhhB,
                     const float* __restrict__ bhhF,
                     const float* __restrict__ bhhB,
                     const __bf16* __restrict__ xpF,
                     const __bf16* __restrict__ xpB,
                     float* __restrict__ hfin) {
    __shared__ float  hF[16 * H_];        // 32 KB  fp32 hidden
    __shared__ __bf16 hB[16 * HP];        // 16.6 KB bf16 hidden (padded rows)
    __shared__ float  Gp[16 * G3];        // 96 KB  gate pre-activations (h-side)

    int dir   = blockIdx.x >> 3;
    int slice = blockIdx.x & 7;
    int b0    = slice * 16;
    const __bf16* PW  = dir ? PWhhB : PWhhF;
    const float*  bhh = dir ? bhhB  : bhhF;
    const __bf16* xp  = dir ? xpB   : xpF;

    int tid  = threadIdx.x;
    int wave = tid >> 5;
    int lane = tid & 31;

    for (int i = tid; i < 16 * H_; i += 512) hF[i] = 0.f;
    for (int i = tid; i < 16 * HP; i += 512) hB[i] = (__bf16)0.f;
    __syncthreads();

    float bias[6];                         // b_hh column for each owned N-tile
#pragma unroll
    for (int q = 0; q < 6; q++)
        bias[q] = bhh[(wave * 6 + q) * 16 + (lane & 15)];

    int arow = lane & 15;
    int koff = (lane < 16) ? 0 : 8;
    int c0   = lane * 16;                  // combine-phase column chunk

    // wave's weight base: fragment (wave*6+j, kc) at elem ((j*16)+kc)*512
    const __bf16* wb = PW + (size_t)(wave * 6) * 16 * 512 + lane * 16;

    for (int t = 0; t < T_; t++) {
        // ---- phase 1: hg = h @ W_hh^T (+ b_hh); 6 chains, one K-sweep
        AccF acc[6];
#pragma unroll
        for (int j = 0; j < 6; j++)
#pragma unroll
            for (int i = 0; i < 8; i++) acc[j].f[i] = 0.f;

#pragma unroll
        for (int kc = 0; kc < 16; kc++) {
            Frag a;
            const __bf16* ap = &hB[arow * HP + kc * 32 + koff];
            a.u[0] = *(const u32x4*)ap;
            a.u[1] = *(const u32x4*)(ap + 16);
#pragma unroll
            for (int j = 0; j < 6; j++) {
                Frag b;
                const u32x4* bp = (const u32x4*)(wb + (size_t)(j * 16 + kc) * 512);
                b.u[0] = bp[0];
                b.u[1] = bp[1];
                acc[j].v = __builtin_amdgcn_wmma_f32_16x16x32_bf16(
                    false, a.v, false, b.v, (short)0, acc[j].v, false, false);
            }
        }
        {
            int rb = (lane < 16) ? 0 : 8;
#pragma unroll
            for (int j = 0; j < 6; j++) {
                int col = (wave * 6 + j) * 16 + (lane & 15);
#pragma unroll
                for (int r = 0; r < 8; r++)
                    Gp[(rb + r) * G3 + col] = acc[j].f[r] + bias[j];
            }
        }
        __syncthreads();

        // ---- phase 2: gate combine; wave = row, lane = 16-column chunk
        {
            int    rowi = wave;
            int    tt   = dir ? (T_ - 1 - t) : t;
            size_t xb   = ((size_t)(b0 + rowi) * T_ + tt) * G3;
            Frag fxr, fxz, fxn;
            const u32x4* pxr = (const u32x4*)(xp + xb + c0);
            const u32x4* pxz = (const u32x4*)(xp + xb + 512 + c0);
            const u32x4* pxn = (const u32x4*)(xp + xb + 1024 + c0);
            fxr.u[0] = __builtin_nontemporal_load(pxr);
            fxr.u[1] = __builtin_nontemporal_load(pxr + 1);
            fxz.u[0] = __builtin_nontemporal_load(pxz);
            fxz.u[1] = __builtin_nontemporal_load(pxz + 1);
            fxn.u[0] = __builtin_nontemporal_load(pxn);
            fxn.u[1] = __builtin_nontemporal_load(pxn + 1);
            const f32x4* pgr = (const f32x4*)&Gp[rowi * G3 + c0];
            const f32x4* pgz = (const f32x4*)&Gp[rowi * G3 + 512 + c0];
            const f32x4* pgn = (const f32x4*)&Gp[rowi * G3 + 1024 + c0];
            f32x4*       phf = (f32x4*)&hF[rowi * H_ + c0];
            Frag hbv;
#pragma unroll
            for (int v4 = 0; v4 < 4; v4++) {
                f32x4 gr = pgr[v4], gz = pgz[v4], gn = pgn[v4], ho = phf[v4];
                f32x4 hn;
#pragma unroll
                for (int e = 0; e < 4; e++) {
                    int   idx  = v4 * 4 + e;
                    float r_   = fsigmoid((float)fxr.h[idx] + gr[e]);
                    float z_   = fsigmoid((float)fxz.h[idx] + gz[e]);
                    float n_   = ftanh((float)fxn.h[idx] + r_ * gn[e]);
                    float h2   = (1.f - z_) * n_ + z_ * ho[e];
                    hn[e]      = h2;
                    hbv.h[idx] = (__bf16)h2;
                }
                phf[v4] = hn;
            }
            u32x4* phb = (u32x4*)&hB[rowi * HP + c0];
            phb[0] = hbv.u[0];
            phb[1] = hbv.u[1];
        }
        __syncthreads();
    }

    for (int rowi = 0; rowi < 16; rowi++)
        hfin[((size_t)dir * B_ + (b0 + rowi)) * H_ + tid] = hF[rowi * H_ + tid];
}

// ---------------------------------------------------------------------------
// Final FC: out[b] = sigmoid( [h_f | h_b] . fc_w + fc_b )
// ---------------------------------------------------------------------------
__global__ __launch_bounds__(256)
void fc_kernel(const float* __restrict__ hfin,
               const float* __restrict__ fcw,
               const float* __restrict__ fcb,
               float* __restrict__ out) {
    __shared__ float red[256];
    int   b   = blockIdx.x;
    int   tid = threadIdx.x;
    float s   = 0.f;
    for (int j = tid; j < 2 * H_; j += 256) {
        float hv = (j < H_) ? hfin[(size_t)b * H_ + j]
                            : hfin[(size_t)(B_ + b) * H_ + (j - H_)];
        s += hv * fcw[j];
    }
    red[tid] = s;
    __syncthreads();
    for (int off = 128; off > 0; off >>= 1) {
        if (tid < off) red[tid] += red[tid + off];
        __syncthreads();
    }
    if (tid == 0) out[b] = fsigmoid(red[0] + fcb[0]);
}

// ---------------------------------------------------------------------------
extern "C" void kernel_launch(void* const* d_in, const int* in_sizes, int n_in,
                              void* d_out, int out_size, void* d_ws, size_t ws_size,
                              hipStream_t stream) {
    const int*   inputs = (const int*)  d_in[0];
    const float* emb    = (const float*)d_in[1];
    const float* W_ih_f = (const float*)d_in[2];
    const float* W_hh_f = (const float*)d_in[3];
    const float* b_ih_f = (const float*)d_in[4];
    const float* b_hh_f = (const float*)d_in[5];
    const float* W_ih_b = (const float*)d_in[6];
    const float* W_hh_b = (const float*)d_in[7];
    const float* b_ih_b = (const float*)d_in[8];
    const float* b_hh_b = (const float*)d_in[9];
    const float* fc_w   = (const float*)d_in[10];
    const float* fc_b   = (const float*)d_in[11];
    float* out = (float*)d_out;

    // Workspace carve-up (256B aligned).
    char*  ws  = (char*)d_ws;
    size_t off = 0;
    auto carve = [&](size_t bytes) {
        char* p = ws + off;
        off += (bytes + 255) & ~(size_t)255;
        return p;
    };
    __bf16* pWihF = (__bf16*)carve((size_t)G3 * E_ * 2);   // 0.79 MB
    __bf16* pWihB = (__bf16*)carve((size_t)G3 * E_ * 2);
    __bf16* pWhhF = (__bf16*)carve((size_t)G3 * H_ * 2);   // 1.57 MB
    __bf16* pWhhB = (__bf16*)carve((size_t)G3 * H_ * 2);
    __bf16* xpF   = (__bf16*)carve((size_t)BT_ * G3 * 2);  // 201 MB
    __bf16* xpB   = (__bf16*)carve((size_t)BT_ * G3 * 2);  // 201 MB
    float*  hfin  = (float*) carve((size_t)2 * B_ * H_ * 4);
    (void)ws_size; (void)in_sizes; (void)n_in; (void)out_size;

    // 1) pack weights into WMMA bf16 B-fragment layout
    {
        int thrIh = (G3 / 16) * (E_ / 32) * 32;   // 24576
        int thrHh = (G3 / 16) * (H_ / 32) * 32;   // 49152
        pack_w_bf16<<<(thrIh + 255) / 256, 256, 0, stream>>>(W_ih_f, pWihF, G3, E_);
        pack_w_bf16<<<(thrIh + 255) / 256, 256, 0, stream>>>(W_ih_b, pWihB, G3, E_);
        pack_w_bf16<<<(thrHh + 255) / 256, 256, 0, stream>>>(W_hh_f, pWhhF, G3, H_);
        pack_w_bf16<<<(thrHh + 255) / 256, 256, 0, stream>>>(W_hh_b, pWhhB, G3, H_);
    }

    // 2) gather + input projection GEMM (both directions)
    xproj_kernel<<<BT_ / 16, 256, 0, stream>>>(inputs, emb, pWihF, pWihB,
                                               b_ih_f, b_ih_b, xpF, xpB);

    // 3) recurrent scan: 16 independent persistent workgroups
    gru_scan_kernel<<<16, 512, 0, stream>>>(pWhhF, pWhhB, b_hh_f, b_hh_b,
                                            xpF, xpB, hfin);

    // 4) final FC + sigmoid
    fc_kernel<<<B_, 256, 0, stream>>>(hfin, fc_w, fc_b, out);
}
